// SelfAttention_66864050864175
// MI455X (gfx1250) — matrix operand order repaired and checked by reading, not compiled, not run on hardware
//
#include <hip/hip_runtime.h>
#include <hip/hip_bf16.h>
#include <math.h>

// ---------------------------------------------------------------------------
// Types
// ---------------------------------------------------------------------------
typedef __attribute__((ext_vector_type(16))) __bf16 v16bf;
typedef __attribute__((ext_vector_type(8)))  __bf16 v8bf;
typedef __attribute__((ext_vector_type(8)))  float  v8f;
typedef __attribute__((ext_vector_type(4)))  unsigned int u32x4;
typedef __attribute__((ext_vector_type(8)))  int i32x8;
typedef __attribute__((ext_vector_type(4)))  int i32x4;

#define B_    8
#define C_    512
#define L_    2048
#define CQK_  64
#define KEYS_PER_STEP 32
#define NSTEPS (L_ / KEYS_PER_STEP)

#if __has_builtin(__builtin_amdgcn_tensor_load_to_lds)
#define HAVE_TDM 1
#else
#define HAVE_TDM 0
#endif

// ---------------------------------------------------------------------------
// Device helpers
// ---------------------------------------------------------------------------
__device__ __forceinline__ v8f vzero8() {
  v8f z;
#pragma unroll
  for (int i = 0; i < 8; ++i) z[i] = 0.0f;
  return z;
}

// D = A(16x32 bf16) * B(32x16 bf16) + C(16x16 f32)
__device__ __forceinline__ v8f wmma_bf16(v16bf a, v16bf b, v8f c) {
  return __builtin_amdgcn_wmma_f32_16x16x32_bf16(false, a, false, b,
                                                 (short)0, c, false, false);
}

// A-matrix (16x32, 16-bit) per-lane load from a row-major row pointer.
// Lane (m = lane&15, half = lane>>4) holds K = {half*16+0..15}.
__device__ __forceinline__ v16bf load_a16(const __bf16* row, int k0, int half) {
  const v8bf lo = *(const v8bf*)(row + k0 + half * 16);
  const v8bf hi = *(const v8bf*)(row + k0 + half * 16 + 8);
  return __builtin_shufflevector(lo, hi, 0,1,2,3,4,5,6,7,8,9,10,11,12,13,14,15);
}

// B-matrix (32x16, 16-bit) per-lane load: lane n holds 16 contiguous K elements
// of column n.  p must point at  colbase + (half?16:0)  and be 32B aligned.
__device__ __forceinline__ v16bf load_b16(const __bf16* p) {
  return *(const v16bf*)p;
}

#if HAVE_TDM
// Tensor Data Mover: 2D tile (tile_d0 contiguous elems x tile_d1 rows) from
// global (row stride stride0, elements) into LDS, packed row-major, 2B elems.
__device__ __forceinline__ void tdm_load_2d(unsigned lds_off, const void* gptr,
                                            unsigned tensor_d0, unsigned tensor_d1,
                                            unsigned tile_d0, unsigned tile_d1,
                                            unsigned long long stride0) {
  unsigned long long ga = (unsigned long long)gptr;
  u32x4 g0;
  g0[0] = 1u;                                   // count=1, user descriptor
  g0[1] = lds_off;                              // LDS byte address
  g0[2] = (unsigned)(ga & 0xffffffffu);         // global_addr[31:0]
  g0[3] = (unsigned)((ga >> 32) & 0x01ffffffu) | (2u << 30);  // addr[56:32]|type=2
  i32x8 g1;
  g1[0] = (int)(1u << 16);                      // wg_mask=0, data_size=1 (2B)
  g1[1] = (int)((tensor_d0 & 0xffffu) << 16);   // tensor_dim0[15:0]
  g1[2] = (int)((tensor_d0 >> 16) | ((tensor_d1 & 0xffffu) << 16));
  g1[3] = (int)((tensor_d1 >> 16) | (tile_d0 << 16));
  g1[4] = (int)(tile_d1 & 0xffffu);             // tile_dim1 (tile_dim2 = 0)
  g1[5] = (int)(unsigned)(stride0 & 0xffffffffu);
  g1[6] = (int)(unsigned)((stride0 >> 32) & 0xffffu);
  g1[7] = 0;
  i32x4 z4; z4[0]=0; z4[1]=0; z4[2]=0; z4[3]=0;
#if __clang_major__ >= 23
  i32x8 z8;
#pragma unroll
  for (int i = 0; i < 8; ++i) z8[i] = 0;
  __builtin_amdgcn_tensor_load_to_lds(g0, g1, z4, z4, z8, 0);
#else
  __builtin_amdgcn_tensor_load_to_lds(g0, g1, z4, z4, 0);
#endif
}
#endif

// ---------------------------------------------------------------------------
// Kernel 1: f32 -> bf16 cast (weights)
// ---------------------------------------------------------------------------
__global__ __launch_bounds__(256) void k_castbf(const float* __restrict__ in,
                                                __bf16* __restrict__ out, int n) {
  int i = blockIdx.x * 256 + threadIdx.x;
  if (i < n) out[i] = (__bf16)in[i];
}

// ---------------------------------------------------------------------------
// Kernel 2: x[b][c][l] f32 -> xb[b][l][c] bf16  (tiled LDS transpose)
// ---------------------------------------------------------------------------
__global__ __launch_bounds__(256) void k_xpose(const float* __restrict__ x,
                                               __bf16* __restrict__ xb) {
  __shared__ float t[32][33];
  const int b  = blockIdx.z;
  const int c0 = blockIdx.y * 32;
  const int l0 = blockIdx.x * 32;
  const int tx = threadIdx.x & 31;
  const int ty = threadIdx.x >> 5;          // 0..7
#pragma unroll
  for (int i = 0; i < 4; ++i)
    t[ty + i * 8][tx] = x[((size_t)b * C_ + c0 + ty + i * 8) * L_ + l0 + tx];
  __syncthreads();
#pragma unroll
  for (int i = 0; i < 4; ++i)
    xb[((size_t)b * L_ + l0 + ty + i * 8) * C_ + c0 + tx] = (__bf16)t[tx][ty + i * 8];
}

// ---------------------------------------------------------------------------
// Kernel 3: Q/K projection.  Out[b][l][o] = xb[b][l][:] . W[o][:] + bias[o]
// One wave per 32-position (2x16) tile; 4 n-tiles each for Q and K.
// Software-pipelined (depth 1): loads for step k+1 issued before step-k WMMAs.
// ---------------------------------------------------------------------------
__global__ __launch_bounds__(256) void k_qkproj(const __bf16* __restrict__ xb,
                                                const __bf16* __restrict__ Wq,
                                                const float*  __restrict__ bq,
                                                const __bf16* __restrict__ Wk,
                                                const float*  __restrict__ bk,
                                                __bf16* __restrict__ Qb,
                                                __bf16* __restrict__ Kb) {
  const int wave = threadIdx.x >> 5, lane = threadIdx.x & 31;
  const int half = lane >> 4, l16 = lane & 15;
  const int wg = blockIdx.x * 8 + wave;       // 0..511
  const int b  = wg >> 6;
  const int l0 = (wg & 63) * 32;
  const __bf16* Arow0 = xb + ((size_t)b * L_ + l0 + l16) * C_;
  const __bf16* Arow1 = Arow0 + (size_t)16 * C_;

  v8f accQ[2][4], accK[2][4];
#pragma unroll
  for (int t = 0; t < 2; ++t)
#pragma unroll
    for (int n = 0; n < 4; ++n) { accQ[t][n] = vzero8(); accK[t][n] = vzero8(); }

  v16bf a0 = load_a16(Arow0, 0, half);
  v16bf a1 = load_a16(Arow1, 0, half);
  v16bf wqv[4], wkv[4];
#pragma unroll
  for (int n = 0; n < 4; ++n) {
    wqv[n] = load_b16(Wq + (size_t)(n * 16 + l16) * C_ + half * 16);
    wkv[n] = load_b16(Wk + (size_t)(n * 16 + l16) * C_ + half * 16);
  }

#pragma unroll
  for (int k0 = 0; k0 < C_; k0 += 32) {
    const int kn = (k0 + 32 < C_) ? k0 + 32 : k0;     // clamp: last prefetch is dead
    v16bf a0n = load_a16(Arow0, kn, half);
    v16bf a1n = load_a16(Arow1, kn, half);
    v16bf wqn[4], wkn[4];
#pragma unroll
    for (int n = 0; n < 4; ++n) {
      wqn[n] = load_b16(Wq + (size_t)(n * 16 + l16) * C_ + kn + half * 16);
      wkn[n] = load_b16(Wk + (size_t)(n * 16 + l16) * C_ + kn + half * 16);
    }
#pragma unroll
    for (int n = 0; n < 4; ++n) {
      accQ[0][n] = wmma_bf16(a0, wqv[n], accQ[0][n]);
      accQ[1][n] = wmma_bf16(a1, wqv[n], accQ[1][n]);
      accK[0][n] = wmma_bf16(a0, wkv[n], accK[0][n]);
      accK[1][n] = wmma_bf16(a1, wkv[n], accK[1][n]);
    }
    a0 = a0n; a1 = a1n;
#pragma unroll
    for (int n = 0; n < 4; ++n) { wqv[n] = wqn[n]; wkv[n] = wkn[n]; }
  }

#pragma unroll
  for (int t = 0; t < 2; ++t) {
#pragma unroll
    for (int n = 0; n < 4; ++n) {
      const float biq = bq[n * 16 + l16];
      const float bik = bk[n * 16 + l16];
#pragma unroll
      for (int r = 0; r < 8; ++r) {
        const size_t row = (size_t)b * L_ + l0 + t * 16 + r + half * 8;
        Qb[row * CQK_ + n * 16 + l16] = (__bf16)(accQ[t][n][r] + biq);
        Kb[row * CQK_ + n * 16 + l16] = (__bf16)(accK[t][n][r] + bik);
      }
    }
  }
}

// ---------------------------------------------------------------------------
// Kernels 4/6 core: channel-major GEMM, 4 position tiles per wave,
// depth-2 software pipeline.
//   acc[j] (16x16) = W[ot*16..][:] . In[b][l0+16j..][:]^T
// ---------------------------------------------------------------------------
__device__ __forceinline__ void chmajor_gemm(const __bf16* __restrict__ W,
                                             const __bf16* __restrict__ In,
                                             int b, int ot, int l0,
                                             int half, int l16, v8f acc[4]) {
  const __bf16* Arow = W + (size_t)(ot * 16 + l16) * C_;
  const __bf16* Bc0 = In + ((size_t)b * L_ + l0 +  0 + l16) * C_;
  const __bf16* Bc1 = In + ((size_t)b * L_ + l0 + 16 + l16) * C_;
  const __bf16* Bc2 = In + ((size_t)b * L_ + l0 + 32 + l16) * C_;
  const __bf16* Bc3 = In + ((size_t)b * L_ + l0 + 48 + l16) * C_;
#pragma unroll
  for (int j = 0; j < 4; ++j) acc[j] = vzero8();

  v16bf aS[2];
  v16bf bS[2][4];
  aS[0] = load_a16(Arow, 0, half);
  bS[0][0] = load_b16(Bc0 + half * 16);
  bS[0][1] = load_b16(Bc1 + half * 16);
  bS[0][2] = load_b16(Bc2 + half * 16);
  bS[0][3] = load_b16(Bc3 + half * 16);
  aS[1] = load_a16(Arow, 32, half);
  bS[1][0] = load_b16(Bc0 + 32 + half * 16);
  bS[1][1] = load_b16(Bc1 + 32 + half * 16);
  bS[1][2] = load_b16(Bc2 + 32 + half * 16);
  bS[1][3] = load_b16(Bc3 + 32 + half * 16);

  int s = 0;
#pragma unroll
  for (int k0 = 0; k0 < C_; k0 += 32) {
    int kp = k0 + 64; if (kp >= C_) kp = C_ - 32;     // clamped dead prefetch
    v16bf an = load_a16(Arow, kp, half);
    v16bf bn0 = load_b16(Bc0 + kp + half * 16);
    v16bf bn1 = load_b16(Bc1 + kp + half * 16);
    v16bf bn2 = load_b16(Bc2 + kp + half * 16);
    v16bf bn3 = load_b16(Bc3 + kp + half * 16);
    acc[0] = wmma_bf16(aS[s], bS[s][0], acc[0]);
    acc[1] = wmma_bf16(aS[s], bS[s][1], acc[1]);
    acc[2] = wmma_bf16(aS[s], bS[s][2], acc[2]);
    acc[3] = wmma_bf16(aS[s], bS[s][3], acc[3]);
    aS[s] = an;
    bS[s][0] = bn0; bS[s][1] = bn1; bS[s][2] = bn2; bS[s][3] = bn3;
    s ^= 1;
  }
}

// Kernel 4: Vc[b][o][l] = Wv[o][:] . xb[b][l][:] + bv[o]   (bf16, ch-major)
__global__ __launch_bounds__(256) void k_vproj(const __bf16* __restrict__ W,
                                               const float*  __restrict__ bias,
                                               const __bf16* __restrict__ In,
                                               __bf16* __restrict__ Out) {
  const int wave = threadIdx.x >> 5, lane = threadIdx.x & 31;
  const int half = lane >> 4, l16 = lane & 15;
  const int wg = blockIdx.x * 8 + wave;       // 0..8191
  const int b  = wg >> 10;
  const int rem = wg & 1023;
  const int ot = rem >> 5;                    // 0..31
  const int l0 = (rem & 31) * 64;             // 4 position tiles

  v8f acc[4];
  chmajor_gemm(W, In, b, ot, l0, half, l16, acc);

#pragma unroll
  for (int j = 0; j < 4; ++j) {
#pragma unroll
    for (int r = 0; r < 8; ++r) {
      const int ch = ot * 16 + r + half * 8;
      Out[((size_t)b * C_ + ch) * L_ + l0 + 16 * j + l16] =
          (__bf16)(acc[j][r] + bias[ch]);
    }
  }
}

// Kernel 6: y[b][o][l] = Wo[o][:] . Ob[b][l][:] + bo[o] + x[b][o][l]  (f32)
__global__ __launch_bounds__(256) void k_oproj(const __bf16* __restrict__ W,
                                               const float*  __restrict__ bias,
                                               const __bf16* __restrict__ In,
                                               const float*  __restrict__ xres,
                                               float* __restrict__ y) {
  const int wave = threadIdx.x >> 5, lane = threadIdx.x & 31;
  const int half = lane >> 4, l16 = lane & 15;
  const int wg = blockIdx.x * 8 + wave;
  const int b  = wg >> 10;
  const int rem = wg & 1023;
  const int ot = rem >> 5;
  const int l0 = (rem & 31) * 64;

  v8f acc[4];
  chmajor_gemm(W, In, b, ot, l0, half, l16, acc);

#pragma unroll
  for (int j = 0; j < 4; ++j) {
#pragma unroll
    for (int r = 0; r < 8; ++r) {
      const int ch = ot * 16 + r + half * 8;
      const size_t idx = ((size_t)b * C_ + ch) * L_ + l0 + 16 * j + l16;
      y[idx] = acc[j][r] + bias[ch] + xres[idx];
    }
  }
}

// ---------------------------------------------------------------------------
// Kernel 5: fused flash attention.
//   8 waves/WG; wave = (qtile 0..3) x (channel-half 0..1); 64 queries per WG.
//   32-key steps; K/V tiles double-buffered in LDS via Tensor Data Mover.
//   Ob[b][q][c] = softmax(Q K^T * scale) . V^T        (bf16 out)
// ---------------------------------------------------------------------------
__global__ __launch_bounds__(256) void k_flash(const __bf16* __restrict__ Qb,
                                               const __bf16* __restrict__ Kb,
                                               const __bf16* __restrict__ Vc,
                                               __bf16* __restrict__ Ob) {
  __shared__ alignas(128) __bf16 Vlds[2][C_][KEYS_PER_STEP];          // 2x32KB
  __shared__ alignas(128) __bf16 Klds[2][KEYS_PER_STEP][CQK_];        // 2x4KB
  __shared__ alignas(128) __bf16 Plds[8][16][KEYS_PER_STEP];          // 8x1KB

  const int tid  = threadIdx.x;
  const int wave = tid >> 5, lane = tid & 31;
  const int half = lane >> 4, l16 = lane & 15;
  const int b    = blockIdx.y;
  const int q0   = blockIdx.x * 64 + (wave >> 1) * 16;
  const int cbase = (wave & 1) * 256;           // this wave's 256 channels
  const float scale = 0.125f;                   // CQK^-0.5

  // Q tile (16 queries x 64 channels) as two A-matrices, resident in VGPRs.
  const __bf16* Qrow = Qb + ((size_t)b * L_ + q0 + l16) * CQK_;
  const v16bf aq0 = load_a16(Qrow, 0, half);
  const v16bf aq1 = load_a16(Qrow, 32, half);

  float m_st[8], l_st[8];
  v8f O[16];
#pragma unroll
  for (int r = 0; r < 8; ++r) { m_st[r] = -INFINITY; l_st[r] = 0.0f; }
#pragma unroll
  for (int t = 0; t < 16; ++t) O[t] = vzero8();

#if HAVE_TDM
  if (wave == 0) {  // preload step 0 into buffer 0
    tdm_load_2d((unsigned)(uintptr_t)&Klds[0][0][0],
                Kb + ((size_t)b * L_) * CQK_,
                CQK_, KEYS_PER_STEP, CQK_, KEYS_PER_STEP, CQK_);
    tdm_load_2d((unsigned)(uintptr_t)&Vlds[0][0][0],
                Vc + (size_t)b * C_ * L_,
                KEYS_PER_STEP, C_, KEYS_PER_STEP, C_, L_);
  }
#endif

  for (int it = 0; it < NSTEPS; ++it) {
#if HAVE_TDM
    const int cur = it & 1;
    if (wave == 0) __builtin_amdgcn_s_wait_tensorcnt(0);
    __syncthreads();                       // buf[cur] visible to all waves
    if (wave == 0 && it + 1 < NSTEPS) {    // prefetch next step (overlapped)
      const int j = (it + 1) * KEYS_PER_STEP, nxt = cur ^ 1;
      tdm_load_2d((unsigned)(uintptr_t)&Klds[nxt][0][0],
                  Kb + ((size_t)b * L_ + j) * CQK_,
                  CQK_, KEYS_PER_STEP, CQK_, KEYS_PER_STEP, CQK_);
      tdm_load_2d((unsigned)(uintptr_t)&Vlds[nxt][0][0],
                  Vc + (size_t)b * C_ * L_ + j,
                  KEYS_PER_STEP, C_, KEYS_PER_STEP, C_, L_);
    }
#else
    const int cur = 0;
    const int j = it * KEYS_PER_STEP;
    for (int idx = tid; idx < C_ * 16; idx += 256) {        // V tile: 512x64B
      const int c = idx >> 4, u = idx & 15;
      ((unsigned*)&Vlds[0][c][0])[u] =
          ((const unsigned*)(Vc + ((size_t)b * C_ + c) * L_ + j))[u];
    }
    for (int idx = tid; idx < KEYS_PER_STEP * 32; idx += 256) {  // K tile
      const int k = idx >> 5, u = idx & 31;
      ((unsigned*)&Klds[0][k][0])[u] =
          ((const unsigned*)(Kb + ((size_t)b * L_ + j + k) * CQK_))[u];
    }
    __syncthreads();
#endif

    // ----- S = Q . K^T for 32 keys (two 16-key N-subtiles), K-depth 64 -----
    // Load all four B tiles up front, then issue the four WMMAs.
    const v16bf bk00 = load_b16(&Klds[cur][l16][half * 16]);
    const v16bf bk01 = load_b16(&Klds[cur][l16][32 + half * 16]);
    const v16bf bk10 = load_b16(&Klds[cur][16 + l16][half * 16]);
    const v16bf bk11 = load_b16(&Klds[cur][16 + l16][32 + half * 16]);
    v8f S0 = vzero8(), S1 = vzero8();
    S0 = wmma_bf16(aq0, bk00, S0);
    S0 = wmma_bf16(aq1, bk01, S0);
    S1 = wmma_bf16(aq0, bk10, S1);
    S1 = wmma_bf16(aq1, bk11, S1);

    // ----- online softmax (f32); row = r + half*8, spread over 16 lanes -----
    float alpha[8];
#pragma unroll
    for (int r = 0; r < 8; ++r) {
      float mx = fmaxf(S0[r], S1[r]);
      mx = fmaxf(mx, __shfl_xor(mx, 1));
      mx = fmaxf(mx, __shfl_xor(mx, 2));
      mx = fmaxf(mx, __shfl_xor(mx, 4));
      mx = fmaxf(mx, __shfl_xor(mx, 8));
      const float mnew = fmaxf(m_st[r], mx * scale);
      alpha[r] = __expf(m_st[r] - mnew);
      const float p0 = __expf(S0[r] * scale - mnew);
      const float p1 = __expf(S1[r] * scale - mnew);
      float rs = p0 + p1;
      rs += __shfl_xor(rs, 1);
      rs += __shfl_xor(rs, 2);
      rs += __shfl_xor(rs, 4);
      rs += __shfl_xor(rs, 8);
      l_st[r] = l_st[r] * alpha[r] + rs;
      m_st[r] = mnew;
      S0[r] = p0;
      S1[r] = p1;
    }

    // ----- restage P (C/D layout) into A layout through wave-private LDS ----
#pragma unroll
    for (int r = 0; r < 8; ++r) {
      Plds[wave][r + half * 8][l16]      = (__bf16)S0[r];
      Plds[wave][r + half * 8][16 + l16] = (__bf16)S1[r];
    }

    // O-rescale (pure VALU) overlaps the DS round-trip latency above.
#pragma unroll
    for (int t = 0; t < 16; ++t) {
#pragma unroll
      for (int r = 0; r < 8; ++r) O[t][r] *= alpha[r];
    }

    asm volatile("s_wait_dscnt 0" ::: "memory");   // wave-internal DS ordering
    const v16bf ap = load_a16(&Plds[wave][l16][0], 0, half);

    // ----- accumulate O += P . V, prefetching the next LDS B tile ----------
    v16bf bv = load_b16(&Vlds[cur][cbase + l16][half * 16]);
#pragma unroll
    for (int t = 0; t < 16; ++t) {
      v16bf bvn = bv;
      if (t < 15)
        bvn = load_b16(&Vlds[cur][cbase + 16 * (t + 1) + l16][half * 16]);
      O[t] = wmma_bf16(ap, bv, O[t]);
      bv = bvn;
    }
    __syncthreads();                       // done with buf[cur]
  }

  // ----- normalize and store Ob[b][q][c] (bf16, lane-contiguous) -----------
  float rl[8];
#pragma unroll
  for (int r = 0; r < 8; ++r) rl[r] = 1.0f / l_st[r];
#pragma unroll
  for (int t = 0; t < 16; ++t) {
#pragma unroll
    for (int r = 0; r < 8; ++r) {
      Ob[((size_t)b * L_ + q0 + r + half * 8) * C_ + cbase + 16 * t + l16] =
          (__bf16)(O[t][r] * rl[r]);
    }
  }
}

// ---------------------------------------------------------------------------
// Host launcher
// ---------------------------------------------------------------------------
extern "C" void kernel_launch(void* const* d_in, const int* in_sizes, int n_in,
                              void* d_out, int out_size, void* d_ws, size_t ws_size,
                              hipStream_t stream) {
  (void)in_sizes; (void)n_in; (void)out_size; (void)ws_size;
  const float* x  = (const float*)d_in[0];
  const float* Wq = (const float*)d_in[1];
  const float* bq = (const float*)d_in[2];
  const float* Wk = (const float*)d_in[3];
  const float* bk = (const float*)d_in[4];
  const float* Wv = (const float*)d_in[5];
  const float* bv = (const float*)d_in[6];
  const float* Wo = (const float*)d_in[7];
  const float* bo = (const float*)d_in[8];
  float* y = (float*)d_out;

  char* ws = (char*)d_ws;
  size_t off = 0;
  __bf16* Wq_b = (__bf16*)(ws + off); off += (size_t)CQK_ * C_ * 2;   // 64 KB
  __bf16* Wk_b = (__bf16*)(ws + off); off += (size_t)CQK_ * C_ * 2;   // 64 KB
  __bf16* Wv_b = (__bf16*)(ws + off); off += (size_t)C_ * C_ * 2;     // 512 KB
  __bf16* Wo_b = (__bf16*)(ws + off); off += (size_t)C_ * C_ * 2;     // 512 KB
  __bf16* xb   = (__bf16*)(ws + off); off += (size_t)B_ * L_ * C_ * 2;   // 16 MB
  __bf16* Qb   = (__bf16*)(ws + off); off += (size_t)B_ * L_ * CQK_ * 2; // 2 MB
  __bf16* Kb   = (__bf16*)(ws + off); off += (size_t)B_ * L_ * CQK_ * 2; // 2 MB
  __bf16* Vc   = (__bf16*)(ws + off); off += (size_t)B_ * C_ * L_ * 2;   // 16 MB
  __bf16* Ob   = (__bf16*)(ws + off); off += (size_t)B_ * L_ * C_ * 2;   // 16 MB

  // 1. weights -> bf16
  k_castbf<<<(CQK_ * C_ + 255) / 256, 256, 0, stream>>>(Wq, Wq_b, CQK_ * C_);
  k_castbf<<<(CQK_ * C_ + 255) / 256, 256, 0, stream>>>(Wk, Wk_b, CQK_ * C_);
  k_castbf<<<(C_ * C_ + 255) / 256, 256, 0, stream>>>(Wv, Wv_b, C_ * C_);
  k_castbf<<<(C_ * C_ + 255) / 256, 256, 0, stream>>>(Wo, Wo_b, C_ * C_);
  // 2. transpose+cast x
  k_xpose<<<dim3(L_ / 32, C_ / 32, B_), 256, 0, stream>>>(x, xb);
  // 3. Q,K projection (2 position tiles / wave)
  k_qkproj<<<(B_ * (L_ / 32)) / 8, 256, 0, stream>>>(xb, Wq_b, bq, Wk_b, bk, Qb, Kb);
  // 4. V projection (channel-major, 4 position tiles / wave)
  k_vproj<<<(B_ * (C_ / 16) * (L_ / 64)) / 8, 256, 0, stream>>>(Wv_b, bv, xb, Vc);
  // 5. fused flash attention
  k_flash<<<dim3(L_ / 64, B_), 256, 0, stream>>>(Qb, Kb, Vc, Ob);
  // 6. output projection + residual
  k_oproj<<<(B_ * (C_ / 16) * (L_ / 64)) / 8, 256, 0, stream>>>(Wo_b, bo, Ob, x, y);
}
